// GINModel_5557687681838
// MI455X (gfx1250) — matrix operand order repaired and checked by reading, not compiled, not run on hardware
//
#include <hip/hip_runtime.h>
#include <hip/hip_bf16.h>

typedef __attribute__((ext_vector_type(16))) __bf16 v16bf;
typedef __attribute__((ext_vector_type(8)))  float  v8f;

#define HID 128
#define NCAT 9
#define PACKED_ELEMS (4 * 128 * 2 * 16)   // kc(4) x col(128) x lanegroup(2) x half(16) = 16384

__device__ __forceinline__ __bf16 f32_to_bf16(float f) {
    unsigned u = __builtin_bit_cast(unsigned, f);
    unsigned r = u + 0x7FFFu + ((u >> 16) & 1u);   // round-to-nearest-even
    unsigned short h = (unsigned short)(r >> 16);
    return __builtin_bit_cast(__bf16, h);
}

// ---------------------------------------------------------------------------
// Repack fp32 W[d_in x 128] (row-major, K-major) into bf16 B-fragment order.
// packed index i = ((kc*128 + n)*2 + g)*16 + j  maps to  K = kc*32 + g*8 + j (j<8), kc*32 + 16 + g*8 + (j-8) (j>=8)
// (matches the A fragment K order: lane group g, halves j<8 and j>=8)
// Rows K >= d_in are zero-padded (layer 0 has K=72).
// ---------------------------------------------------------------------------
__global__ void pack_weights(const float* __restrict__ w, __bf16* __restrict__ out, int d_in) {
    int i = blockIdx.x * 256 + threadIdx.x;
    if (i >= PACKED_ELEMS) return;
    int j  = i & 15;
    int g  = (i >> 4) & 1;
    int n  = (i >> 5) & 127;
    int kc = i >> 12;
    int k  = kc * 32 + (j < 8 ? g * 8 + j : 16 + g * 8 + (j - 8));
    float v = (k < d_in) ? w[(long)k * HID + n] : 0.0f;
    out[i] = f32_to_bf16(v);
}

// ---------------------------------------------------------------------------
// Categorical embedding gather -> h[N x 128], cols 72..127 zeroed.
// ---------------------------------------------------------------------------
__global__ void embed_kernel(const int* __restrict__ x,
                             const float* __restrict__ e0, const float* __restrict__ e1,
                             const float* __restrict__ e2, const float* __restrict__ e3,
                             const float* __restrict__ e4, const float* __restrict__ e5,
                             const float* __restrict__ e6, const float* __restrict__ e7,
                             const float* __restrict__ e8,
                             float* __restrict__ h, int n_nodes) {
    long i = (long)blockIdx.x * 256 + threadIdx.x;
    if (i >= (long)n_nodes * HID) return;
    int node = (int)(i >> 7);
    int c    = (int)(i & 127);
    float v = 0.0f;
    if (c < NCAT * 8) {
        const float* tabs[NCAT] = {e0, e1, e2, e3, e4, e5, e6, e7, e8};
        int cat = c >> 3, e = c & 7;
        int idx = x[(long)node * NCAT + cat];
        v = tabs[cat][(long)idx * 8 + e];
    }
    h[i] = v;
}

__global__ void zero_f32x4(float4* __restrict__ p, long n4) {
    long i = (long)blockIdx.x * 256 + threadIdx.x;
    if (i < n4) p[i] = make_float4(0.f, 0.f, 0.f, 0.f);
}

// ---------------------------------------------------------------------------
// Edge aggregation: one wave per edge. Lane l handles 4 consecutive floats.
// agg[dst] += h[src]   (float4 gather + 4x global_atomic_add_f32 scatter)
// ---------------------------------------------------------------------------
__global__ void edge_agg_kernel(const int* __restrict__ src, const int* __restrict__ dst,
                                const float* __restrict__ h, float* __restrict__ agg,
                                int n_edges) {
    long t = (long)blockIdx.x * 256 + threadIdx.x;
    long e = t >> 5;
    if (e >= n_edges) return;
    int lane = (int)(t & 31);
    int s = src[e], d = dst[e];
    float4 hv = *(const float4*)(h + (long)s * HID + lane * 4);
    float* ap = agg + (long)d * HID + lane * 4;
    atomicAdd(ap + 0, hv.x);
    atomicAdd(ap + 1, hv.y);
    atomicAdd(ap + 2, hv.z);
    atomicAdd(ap + 3, hv.w);
}

// ---------------------------------------------------------------------------
// Fused GIN MLP for a 16-node tile: z=(h+agg); t1=relu(z@W1+b1); out=t1@W2+b2
// 256 threads = 8 waves; wave w owns output columns [16w,16w+16).
// A fragments from LDS per ISA 16-bit A 16x32 layout:
//   lane group g = lane>>4, halves j<8 -> K = kc*32 + g*8 + j
//                            j>=8      -> K = kc*32 + 16 + g*8 + (j-8)
// B fragments are contiguous 32B per lane in the packed weight buffer.
// In-place safe: block reads only its own 16 rows before writing them.
// ---------------------------------------------------------------------------
__global__ void __launch_bounds__(256)
gin_mlp_kernel(const float* __restrict__ hin, const float* __restrict__ agg,
               const __bf16* __restrict__ w1p, const float* __restrict__ b1,
               const __bf16* __restrict__ w2p, const float* __restrict__ b2,
               float* __restrict__ hout, int n_nodes, int relu_out) {
    __shared__ __bf16 zt[16 * HID];
    __shared__ __bf16 t1[16 * HID];

    const int node0 = blockIdx.x * 16;
    const int tid   = threadIdx.x;
    const int wave  = tid >> 5;
    const int lane  = tid & 31;
    const int g     = lane >> 4;
    const int l15   = lane & 15;
    const int col   = wave * 16 + l15;

    // Stage 0: z tile -> LDS as bf16 (row-major 16x128)
    for (int i = tid; i < 16 * HID; i += 256) {
        int r = i >> 7, c = i & 127;
        int node = node0 + r;
        float z = 0.0f;
        if (node < n_nodes) {
            long off = (long)node * HID + c;
            z = hin[off] + agg[off];
        }
        zt[i] = f32_to_bf16(z);
    }
    __syncthreads();

    // Stage 1: t1 = relu(z @ W1 + b1)
    v8f acc = {};
    #pragma unroll
    for (int kc = 0; kc < 4; ++kc) {
        union { v16bf v; uint4 q[2]; } af;
        const __bf16* ar = &zt[l15 * HID + kc * 32 + g * 8];
        af.q[0] = *(const uint4*)(ar);        // halves 0..7  : K = kc*32+g*8 ..+7
        af.q[1] = *(const uint4*)(ar + 16);   // halves 8..15 : K = kc*32+16+g*8 ..+7
        union { v16bf v; uint4 q[2]; } bf_;
        const __bf16* bp = w1p + ((((long)kc * 128 + col) * 2 + g) * 16);
        bf_.q[0] = *(const uint4*)(bp);
        bf_.q[1] = *(const uint4*)(bp + 8);
        acc = __builtin_amdgcn_wmma_f32_16x16x32_bf16(false, af.v, false, bf_.v,
                                                      (short)0, acc, false, false);
    }
    {
        float bias = b1[col];
        #pragma unroll
        for (int r = 0; r < 8; ++r) {
            int m = r + g * 8;                         // C/D: vgpr r -> row r + 8*(lane>=16)
            float v = fmaxf(acc[r] + bias, 0.0f);
            t1[m * HID + col] = f32_to_bf16(v);
        }
    }
    __syncthreads();

    // Stage 2: out = t1 @ W2 + b2 (+ optional relu)
    v8f acc2 = {};
    #pragma unroll
    for (int kc = 0; kc < 4; ++kc) {
        union { v16bf v; uint4 q[2]; } af;
        const __bf16* ar = &t1[l15 * HID + kc * 32 + g * 8];
        af.q[0] = *(const uint4*)(ar);
        af.q[1] = *(const uint4*)(ar + 16);
        union { v16bf v; uint4 q[2]; } bf_;
        const __bf16* bp = w2p + ((((long)kc * 128 + col) * 2 + g) * 16);
        bf_.q[0] = *(const uint4*)(bp);
        bf_.q[1] = *(const uint4*)(bp + 8);
        acc2 = __builtin_amdgcn_wmma_f32_16x16x32_bf16(false, af.v, false, bf_.v,
                                                       (short)0, acc2, false, false);
    }
    {
        float bias = b2[col];
        #pragma unroll
        for (int r = 0; r < 8; ++r) {
            int m = r + g * 8;
            int node = node0 + m;
            if (node < n_nodes) {
                float v = acc2[r] + bias;
                if (relu_out) v = fmaxf(v, 0.0f);
                hout[(long)node * HID + col] = v;
            }
        }
    }
}

// ---------------------------------------------------------------------------
// global_add_pool: one wave per node, atomic add into out[batch[node]]
// ---------------------------------------------------------------------------
__global__ void pool_kernel(const float* __restrict__ h, const int* __restrict__ batch,
                            float* __restrict__ out, int n_nodes) {
    long t = (long)blockIdx.x * 256 + threadIdx.x;
    long node = t >> 5;
    if (node >= n_nodes) return;
    int lane = (int)(t & 31);
    int gph = batch[node];
    const float4 hv = *(const float4*)(h + node * HID + lane * 4);
    float* op = out + (long)gph * HID + lane * 4;
    atomicAdd(op + 0, hv.x);
    atomicAdd(op + 1, hv.y);
    atomicAdd(op + 2, hv.z);
    atomicAdd(op + 3, hv.w);
}

extern "C" void kernel_launch(void* const* d_in, const int* in_sizes, int n_in,
                              void* d_out, int out_size, void* d_ws, size_t ws_size,
                              hipStream_t stream) {
    const int n_nodes = in_sizes[0] / NCAT;       // 100000
    const int n_edges = in_sizes[1] / 2;          // 1600000

    const int* x     = (const int*)d_in[0];
    const int* src   = (const int*)d_in[1];
    const int* dst   = src + n_edges;
    const int* batch = (const int*)d_in[2];
    const float* emb[NCAT];
    for (int i = 0; i < NCAT; ++i) emb[i] = (const float*)d_in[3 + i];
    // mlps flattened after embeddings: per layer {w1, b1, w2, b2}
    const int MLP0 = 3 + NCAT;

    float*  h    = (float*)d_ws;
    float*  agg  = h + (size_t)n_nodes * HID;
    __bf16* wpk  = (__bf16*)(agg + (size_t)n_nodes * HID);   // 10 x 16384 bf16

    // --- pack all weights to bf16 fragment order (tiny, one-time per launch)
    for (int l = 0; l < 5; ++l) {
        const float* w1 = (const float*)d_in[MLP0 + 4 * l + 0];
        const float* w2 = (const float*)d_in[MLP0 + 4 * l + 2];
        int din1 = (l == 0) ? (NCAT * 8) : HID;
        pack_weights<<<PACKED_ELEMS / 256, 256, 0, stream>>>(w1, wpk + (size_t)(2 * l) * PACKED_ELEMS, din1);
        pack_weights<<<PACKED_ELEMS / 256, 256, 0, stream>>>(w2, wpk + (size_t)(2 * l + 1) * PACKED_ELEMS, HID);
    }

    // --- embedding gather (pads cols 72..127 with zeros)
    {
        long total = (long)n_nodes * HID;
        int blocks = (int)((total + 255) / 256);
        embed_kernel<<<blocks, 256, 0, stream>>>(x, emb[0], emb[1], emb[2], emb[3], emb[4],
                                                 emb[5], emb[6], emb[7], emb[8], h, n_nodes);
    }

    // --- 5 GIN layers
    long n4 = (long)n_nodes * HID / 4;
    int zero_blocks = (int)((n4 + 255) / 256);
    int edge_blocks = (int)(((long)n_edges * 32 + 255) / 256);
    int mlp_blocks  = (n_nodes + 15) / 16;
    for (int l = 0; l < 5; ++l) {
        zero_f32x4<<<zero_blocks, 256, 0, stream>>>((float4*)agg, n4);
        edge_agg_kernel<<<edge_blocks, 256, 0, stream>>>(src, dst, h, agg, n_edges);
        const float* b1 = (const float*)d_in[MLP0 + 4 * l + 1];
        const float* b2 = (const float*)d_in[MLP0 + 4 * l + 3];
        gin_mlp_kernel<<<mlp_blocks, 256, 0, stream>>>(
            h, agg,
            wpk + (size_t)(2 * l) * PACKED_ELEMS, b1,
            wpk + (size_t)(2 * l + 1) * PACKED_ELEMS, b2,
            h, n_nodes, (l < 4) ? 1 : 0);
    }

    // --- global add pool
    {
        long o4 = (long)out_size / 4;
        int ob = (int)((o4 + 255) / 256);
        zero_f32x4<<<ob, 256, 0, stream>>>((float4*)d_out, o4);
        int pb = (int)(((long)n_nodes * 32 + 255) / 256);
        pool_kernel<<<pb, 256, 0, stream>>>(h, batch, (float*)d_out, n_nodes);
    }
}
